// FusedMoE_84104049590646
// MI455X (gfx1250) — matrix operand order repaired and checked by reading, not compiled, not run on hardware
//
// FusedMoE (SwiGLU, top-2 of 8 experts) for MI455X / gfx1250.
//
// Roofline: sparse work = 103 GFLOP; weights = 201 MB fp32 (~9 us @ 23.3 TB/s).
// fp32 WMMA (16x16x4) is 8x slower than bf16 WMMA (16x16x32), so we convert
// inputs to bf16 on the fly and accumulate in f32 via v_wmma_f32_16x16x32_bf16,
// landing near the bandwidth/compute balance point instead of 8x compute-bound.
//
// Pipeline: zero_out -> moe_route (expert compaction, BM-padded segments)
//        -> moe_gemm1 (gather X, X@wi0 & X@wi1 shared-A, SiLU*mul, bf16 act)
//        -> moe_gemm2 (act@wo, scale, deterministic f32 atomic add into out).
//
// Round-2 change: preload ALL LDS fragments of a K-step into registers before
// the WMMA chain so the compiler emits one DS clause + partial s_wait_dscnt
// instead of a full `s_wait_dscnt 0` barrier in front of every v_wmma.

#include <hip/hip_runtime.h>
#include <hip/hip_bf16.h>

#define NEXP   8
#define HDIM   1024
#define DDIM   2048
#define NTOK   4096
#define TOPK   2
#define NPAIR  (NTOK * TOPK)          // 8192 (always, regardless of routing)
#define BM     64                     // block tile rows
#define BN     128                    // block tile cols
#define BK     32                     // K-step == WMMA depth
#define SKS    48                     // LDS row stride (bf16 units) = 96B, 16B-aligned
#define ROWCAP (NPAIR + NEXP * BM)    // 8704: worst-case padded compacted rows
#define MBLK   (ROWCAP / BM)          // 136 M-blocks

typedef __bf16 bf16;
typedef __attribute__((ext_vector_type(16))) __bf16 bf16x16;
typedef __attribute__((ext_vector_type(8)))  __bf16 bf16x8;
typedef __attribute__((ext_vector_type(8)))  float  f32x8;

static __device__ __forceinline__ bf16x16 concat16(bf16x8 lo, bf16x8 hi) {
  return __builtin_shufflevector(lo, hi, 0, 1, 2, 3, 4, 5, 6, 7,
                                         8, 9, 10, 11, 12, 13, 14, 15);
}

static __device__ __forceinline__ f32x8 wmma_bf16(bf16x16 a, bf16x16 b, f32x8 c) {
  // (neg_a, A, neg_b, B, c_mod, C, reuse_a, reuse_b)
  return __builtin_amdgcn_wmma_f32_16x16x32_bf16(false, a, false, b,
                                                 (short)0, c, false, false);
}

// ---------------------------------------------------------------------------
// Kernel 0: zero the output (harness does not re-poison between replays).
// ---------------------------------------------------------------------------
__global__ void zero_out_kernel(float4* __restrict__ p, int n4) {
  int i = blockIdx.x * blockDim.x + threadIdx.x;
  if (i < n4) p[i] = make_float4(0.f, 0.f, 0.f, 0.f);
}

// ---------------------------------------------------------------------------
// Kernel 1: routing. One block, 8 waves; wave w compacts pairs of expert w in
// ascending pair order (deterministic). Segment starts padded to BM so every
// GEMM block maps to exactly one expert. Pad slots: pair_row=-1, scale=0.
// ---------------------------------------------------------------------------
__global__ void moe_route(const int* __restrict__ ids,     // [NPAIR]
                          const float* __restrict__ wts,   // [NPAIR]
                          int* __restrict__ pair_row,      // [ROWCAP]
                          float* __restrict__ pair_scale,  // [ROWCAP]
                          int* __restrict__ seg_start) {   // [NEXP+1]
  __shared__ int s_cnt[NEXP];
  __shared__ int s_start[NEXP + 1];
  const int tid = threadIdx.x;
  const int wid = tid >> 5;
  const int lane = tid & 31;
  const unsigned long long ltmask = (1ull << lane) - 1ull;

  int cnt = 0;
  for (int base = 0; base < NPAIR; base += 32) {
    int id = ids[base + lane];
    unsigned long long m = __ballot(id == wid);
    cnt += __popcll(m);
  }
  if (lane == 0) s_cnt[wid] = cnt;
  __syncthreads();
  if (tid == 0) {
    int s = 0;
    for (int e = 0; e < NEXP; ++e) {
      s_start[e] = s;
      s += (s_cnt[e] + BM - 1) / BM * BM;
    }
    s_start[NEXP] = s;
  }
  __syncthreads();
  if (tid <= NEXP) seg_start[tid] = s_start[tid];

  const int base_row = s_start[wid];
  int done = 0;
  for (int base = 0; base < NPAIR; base += 32) {
    int p = base + lane;
    bool hit = (ids[p] == wid);
    unsigned long long m = __ballot(hit);
    if (hit) {
      int pos = done + __popcll(m & ltmask);
      pair_row[base_row + pos] = p;
      pair_scale[base_row + pos] = wts[p];
    }
    done += __popcll(m);
  }
  for (int r = base_row + done + lane; r < s_start[wid + 1]; r += 32) {
    pair_row[r] = -1;
    pair_scale[r] = 0.0f;
  }
}

// ---------------------------------------------------------------------------
// Kernel 2: grouped GEMM1 + SwiGLU.  act[row] = silu(X@wi0) * (X@wi1), bf16.
// Block tile 64x128, 8 waves (4 M-strips x 2 N-halves), each wave 16x64 per
// weight matrix (2x4 accumulators of v8f), K-step 32, double-buffered LDS.
// LDS is fragment-ready: each lane's v16bf = two ds_load_b128 (ISA 7.12.2).
// ---------------------------------------------------------------------------
__global__ __launch_bounds__(256)
void moe_gemm1(const float* __restrict__ x,    // [NTOK, HDIM]
               const float* __restrict__ wi0,  // [NEXP, HDIM, DDIM]
               const float* __restrict__ wi1,  // [NEXP, HDIM, DDIM]
               const int* __restrict__ pair_row,
               const int* __restrict__ seg_start,
               bf16* __restrict__ act) {       // [ROWCAP, DDIM]
  __shared__ __align__(16) bf16 sA[2][BM][SKS];
  __shared__ __align__(16) bf16 sB0[2][BN][SKS];
  __shared__ __align__(16) bf16 sB1[2][BN][SKS];
  __shared__ int s_seg[NEXP + 1];

  const int tid = threadIdx.x;
  if (tid <= NEXP) s_seg[tid] = seg_start[tid];
  __syncthreads();

  const int row0 = blockIdx.x * BM;
  if (row0 >= s_seg[NEXP]) return;  // block-uniform exit
  int e = 0;
  while (row0 >= s_seg[e + 1]) ++e;

  const int n0 = blockIdx.y * BN;
  const float* __restrict__ w0e = wi0 + (size_t)e * HDIM * DDIM;
  const float* __restrict__ w1e = wi1 + (size_t)e * HDIM * DDIM;

  // A gather: thread covers row am, k-quarter aq (8 floats).
  const int am = tid >> 2;
  const int aq = tid & 3;
  const int pr = pair_row[row0 + am];
  const int tok = (pr < 0) ? 0 : (pr >> 1);
  const float* __restrict__ xrow = x + (size_t)tok * HDIM + aq * 8;

  // B column-strip: thread covers column bn, k-half bkh (16 rows of wi).
  const int bn = tid & 127;
  const int bkh = tid >> 7;
  const float* __restrict__ bp0 = w0e + (size_t)(bkh * 16) * DDIM + n0 + bn;
  const float* __restrict__ bp1 = w1e + (size_t)(bkh * 16) * DDIM + n0 + bn;

  // Wave tile mapping.
  const int wid = tid >> 5, wm = wid & 3, wn = wid >> 2;
  const int lane = tid & 31, g = lane >> 4, ln = lane & 15;

  f32x8 acc0[4], acc1[4];
  const f32x8 zero = {0.f, 0.f, 0.f, 0.f, 0.f, 0.f, 0.f, 0.f};
#pragma unroll
  for (int j = 0; j < 4; ++j) { acc0[j] = zero; acc1[j] = zero; }

  float ar[8];
  float b0r[16], b1r[16];

  auto load_step = [&](int s) {
    const int h0 = s * BK;
    float4 av0 = *(const float4*)(xrow + h0);
    float4 av1 = *(const float4*)(xrow + h0 + 4);
    ar[0] = av0.x; ar[1] = av0.y; ar[2] = av0.z; ar[3] = av0.w;
    ar[4] = av1.x; ar[5] = av1.y; ar[6] = av1.z; ar[7] = av1.w;
    const float* p0 = bp0 + (size_t)h0 * DDIM;
    const float* p1 = bp1 + (size_t)h0 * DDIM;
#pragma unroll
    for (int k = 0; k < 16; ++k) {
      b0r[k] = p0[(size_t)k * DDIM];
      b1r[k] = p1[(size_t)k * DDIM];
    }
  };

  auto store_step = [&](int buf) {
    bf16x8 av;
#pragma unroll
    for (int i = 0; i < 8; ++i) av[i] = (bf16)ar[i];
    *(bf16x8*)&sA[buf][am][aq * 8] = av;
    bf16x8 v0l, v0h, v1l, v1h;
#pragma unroll
    for (int i = 0; i < 8; ++i) {
      v0l[i] = (bf16)b0r[i];     v0h[i] = (bf16)b0r[8 + i];
      v1l[i] = (bf16)b1r[i];     v1h[i] = (bf16)b1r[8 + i];
    }
    *(bf16x8*)&sB0[buf][bn][bkh * 16]     = v0l;
    *(bf16x8*)&sB0[buf][bn][bkh * 16 + 8] = v0h;
    *(bf16x8*)&sB1[buf][bn][bkh * 16]     = v1l;
    *(bf16x8*)&sB1[buf][bn][bkh * 16 + 8] = v1h;
  };

  auto compute = [&](int buf) {
    // Issue ALL fragment loads first (single DS clause), then the WMMA chain:
    // lets the scheduler overlap ds_load_b128 latency with matrix ops instead
    // of a full s_wait_dscnt 0 before every wmma.
    const bf16* Ab = &sA[buf][wm * 16 + ln][0];
    bf16x16 aF = concat16(*(const bf16x8*)(Ab + 16 * g),
                          *(const bf16x8*)(Ab + 16 * g + 8));
    bf16x16 bF0[4], bF1[4];
#pragma unroll
    for (int j = 0; j < 4; ++j) {
      const bf16* Bb0 = &sB0[buf][wn * 64 + j * 16 + ln][16 * g];
      bF0[j] = concat16(*(const bf16x8*)Bb0, *(const bf16x8*)(Bb0 + 8));
      const bf16* Bb1 = &sB1[buf][wn * 64 + j * 16 + ln][16 * g];
      bF1[j] = concat16(*(const bf16x8*)Bb1, *(const bf16x8*)(Bb1 + 8));
    }
#pragma unroll
    for (int j = 0; j < 4; ++j) {
      acc0[j] = wmma_bf16(aF, bF0[j], acc0[j]);
      acc1[j] = wmma_bf16(aF, bF1[j], acc1[j]);
    }
  };

  const int STEPS = HDIM / BK;  // 32
  load_step(0);
  store_step(0);
  __syncthreads();
  for (int s = 0; s < STEPS; ++s) {
    const int buf = s & 1;
    if (s + 1 < STEPS) load_step(s + 1);
    compute(buf);
    if (s + 1 < STEPS) store_step(buf ^ 1);
    __syncthreads();
  }

  // Epilogue: SwiGLU, store bf16 activations (C/D layout: M = r + 8*g).
#pragma unroll
  for (int j = 0; j < 4; ++j) {
#pragma unroll
    for (int r = 0; r < 8; ++r) {
      float a0 = acc0[j][r];
      float a1 = acc1[j][r];
      float sig = a0 / (1.0f + __expf(-a0));
      int row = row0 + wm * 16 + g * 8 + r;
      int col = n0 + wn * 64 + j * 16 + ln;
      act[(size_t)row * DDIM + col] = (bf16)(sig * a1);
    }
  }
}

// ---------------------------------------------------------------------------
// Kernel 3: grouped GEMM2 (act @ wo[e]) * routing-weight, atomic f32 add into
// out. Exactly 2 adds per element on a zeroed base -> bitwise deterministic.
// ---------------------------------------------------------------------------
__global__ __launch_bounds__(256)
void moe_gemm2(const bf16* __restrict__ act,   // [ROWCAP, DDIM]
               const float* __restrict__ wo,   // [NEXP, DDIM, HDIM]
               const int* __restrict__ pair_row,
               const float* __restrict__ pair_scale,
               const int* __restrict__ seg_start,
               float* __restrict__ out) {      // [NTOK, HDIM]
  __shared__ __align__(16) bf16 sA[2][BM][SKS];
  __shared__ __align__(16) bf16 sB[2][BN][SKS];
  __shared__ int s_seg[NEXP + 1];

  const int tid = threadIdx.x;
  if (tid <= NEXP) s_seg[tid] = seg_start[tid];
  __syncthreads();

  const int row0 = blockIdx.x * BM;
  if (row0 >= s_seg[NEXP]) return;
  int e = 0;
  while (row0 >= s_seg[e + 1]) ++e;

  const int n0 = blockIdx.y * BN;
  const float* __restrict__ woe = wo + (size_t)e * DDIM * HDIM;

  const int am = tid >> 2;
  const int aq = tid & 3;
  const bf16* __restrict__ arow = act + (size_t)(row0 + am) * DDIM + aq * 8;

  const int bn = tid & 127;
  const int bkh = tid >> 7;
  const float* __restrict__ bp = woe + (size_t)(bkh * 16) * HDIM + n0 + bn;

  const int wid = tid >> 5, wm = wid & 3, wn = wid >> 2;
  const int lane = tid & 31, g = lane >> 4, ln = lane & 15;

  f32x8 acc[4];
  const f32x8 zero = {0.f, 0.f, 0.f, 0.f, 0.f, 0.f, 0.f, 0.f};
#pragma unroll
  for (int j = 0; j < 4; ++j) acc[j] = zero;

  bf16x8 areg;
  float br[16];

  auto load_step = [&](int s) {
    areg = *(const bf16x8*)(arow + s * BK);  // act already bf16
    const float* p = bp + (size_t)(s * BK) * HDIM;
#pragma unroll
    for (int k = 0; k < 16; ++k) br[k] = p[(size_t)k * HDIM];
  };

  auto store_step = [&](int buf) {
    *(bf16x8*)&sA[buf][am][aq * 8] = areg;
    bf16x8 vl, vh;
#pragma unroll
    for (int i = 0; i < 8; ++i) { vl[i] = (bf16)br[i]; vh[i] = (bf16)br[8 + i]; }
    *(bf16x8*)&sB[buf][bn][bkh * 16]     = vl;
    *(bf16x8*)&sB[buf][bn][bkh * 16 + 8] = vh;
  };

  auto compute = [&](int buf) {
    const bf16* Ab = &sA[buf][wm * 16 + ln][0];
    bf16x16 aF = concat16(*(const bf16x8*)(Ab + 16 * g),
                          *(const bf16x8*)(Ab + 16 * g + 8));
    bf16x16 bF[4];
#pragma unroll
    for (int j = 0; j < 4; ++j) {
      const bf16* Bb = &sB[buf][wn * 64 + j * 16 + ln][16 * g];
      bF[j] = concat16(*(const bf16x8*)Bb, *(const bf16x8*)(Bb + 8));
    }
#pragma unroll
    for (int j = 0; j < 4; ++j) acc[j] = wmma_bf16(aF, bF[j], acc[j]);
  };

  const int STEPS = DDIM / BK;  // 64
  load_step(0);
  store_step(0);
  __syncthreads();
  for (int s = 0; s < STEPS; ++s) {
    const int buf = s & 1;
    if (s + 1 < STEPS) load_step(s + 1);
    compute(buf);
    if (s + 1 < STEPS) store_step(buf ^ 1);
    __syncthreads();
  }

  // Epilogue: scale by routing weight, deterministic atomic add into out.
#pragma unroll
  for (int r = 0; r < 8; ++r) {
    const int row = row0 + wm * 16 + g * 8 + r;
    const int prr = pair_row[row];
    if (prr < 0) continue;  // padded slot
    const float sc = pair_scale[row];
    float* __restrict__ orow = out + (size_t)(prr >> 1) * HDIM;
#pragma unroll
    for (int j = 0; j < 4; ++j) {
      const int col = n0 + wn * 64 + j * 16 + ln;
      atomicAdd(orow + col, acc[j][r] * sc);
    }
  }
}

// ---------------------------------------------------------------------------
// Host launcher. Workspace layout (~36 MB):
//   [0,256)                : int seg_start[NEXP+1]
//   [256, +ROWCAP*4)       : int pair_row[ROWCAP]
//   [.., +ROWCAP*4)        : float pair_scale[ROWCAP]
//   [69888, +ROWCAP*DDIM*2): bf16 act[ROWCAP][DDIM]
// ---------------------------------------------------------------------------
extern "C" void kernel_launch(void* const* d_in, const int* in_sizes, int n_in,
                              void* d_out, int out_size, void* d_ws, size_t ws_size,
                              hipStream_t stream) {
  (void)in_sizes; (void)n_in; (void)out_size; (void)ws_size;
  const float* x    = (const float*)d_in[0];   // hidden_states [T,H]
  const float* tw   = (const float*)d_in[1];   // topk_weights  [T,K]
  const int*   tids = (const int*)d_in[2];     // topk_ids      [T,K]
  const float* wi0  = (const float*)d_in[3];   // [E,H,D]
  const float* wi1  = (const float*)d_in[4];   // [E,H,D]
  const float* wo   = (const float*)d_in[5];   // [E,D,H]
  float* out = (float*)d_out;

  char* ws = (char*)d_ws;
  int*   seg_start  = (int*)(ws);
  int*   pair_row   = (int*)(ws + 256);
  float* pair_scale = (float*)(ws + 256 + (size_t)ROWCAP * 4);
  bf16*  act        = (bf16*)(ws + 256 + (size_t)ROWCAP * 8);

  const int n4 = NTOK * HDIM / 4;
  hipLaunchKernelGGL(zero_out_kernel, dim3((n4 + 255) / 256), dim3(256), 0,
                     stream, (float4*)out, n4);
  hipLaunchKernelGGL(moe_route, dim3(1), dim3(256), 0, stream,
                     tids, tw, pair_row, pair_scale, seg_start);
  hipLaunchKernelGGL(moe_gemm1, dim3(MBLK, DDIM / BN), dim3(256), 0, stream,
                     x, wi0, wi1, pair_row, seg_start, act);
  hipLaunchKernelGGL(moe_gemm2, dim3(MBLK, HDIM / BN), dim3(256), 0, stream,
                     act, wo, pair_row, pair_scale, seg_start, out);
}
